// RNNEncoder_54752243089607
// MI455X (gfx1250) — matrix operand order repaired and checked by reading, not compiled, not run on hardware
//
#include <hip/hip_runtime.h>
#include <math.h>

// ---------------------------------------------------------------------------
// Types for CDNA5 WMMA (wave32): v_wmma_f32_16x16x32_bf16
// ---------------------------------------------------------------------------
typedef __bf16 bf16_t;
typedef bf16_t v8bf  __attribute__((ext_vector_type(8)));
typedef bf16_t v16bf __attribute__((ext_vector_type(16)));
typedef float  v8f   __attribute__((ext_vector_type(8)));

static constexpr int Bb = 128;   // batch
static constexpr int Tt = 512;   // timesteps
static constexpr int Dd = 1024;  // input dim
static constexpr int Hh = 1024;  // hidden dim

__device__ __forceinline__ v8bf ldv8(const bf16_t* p) {
    return *reinterpret_cast<const v8bf*>(p);
}
__device__ __forceinline__ v16bf cat8(v8bf lo, v8bf hi) {
    return __builtin_shufflevector(lo, hi,
        0,1,2,3,4,5,6,7,8,9,10,11,12,13,14,15);
}
__device__ __forceinline__ v8f wmma_bf16(v16bf a, v16bf b, v8f c) {
    return __builtin_amdgcn_wmma_f32_16x16x32_bf16(
        false, a, false, b, (short)0, c, false, false);
}

// Async global->LDS copy of 16 bytes per lane (tracked by ASYNCcnt).
// dst: LDS byte address (low 32 bits of the generic pointer to __shared__).
__device__ __forceinline__ void async_copy_b128(void* lds_dst, const void* gsrc) {
    unsigned lds_off = (unsigned)(size_t)lds_dst;
    asm volatile("global_load_async_to_lds_b128 %0, %1, off"
                 :
                 : "v"(lds_off), "v"(gsrc)
                 : "memory");
}
__device__ __forceinline__ void wait_asynccnt0() {
    asm volatile("s_wait_asynccnt 0x0" ::: "memory");
}

// ---------------------------------------------------------------------------
// Small prep kernels
// ---------------------------------------------------------------------------
__global__ void cvt_f32_to_bf16(const float* __restrict__ src,
                                bf16_t* __restrict__ dst, int n) {
    int i = blockIdx.x * blockDim.x + threadIdx.x;
    if (i < n) dst[i] = (bf16_t)src[i];
}

__global__ void bias_sum_kernel(const float* __restrict__ a,
                                const float* __restrict__ b,
                                float* __restrict__ o, int n) {
    int i = blockIdx.x * blockDim.x + threadIdx.x;
    if (i < n) o[i] = a[i] + b[i];
}

__global__ void zero_bf16_kernel(bf16_t* __restrict__ p, int n) {
    int i = blockIdx.x * blockDim.x + threadIdx.x;
    if (i < n) p[i] = (bf16_t)0.0f;
}

// ---------------------------------------------------------------------------
// Phase 1: xp[t][b][h] = x[b,t,:] . W_ih[h,:] + (b_ih+b_hh)[h], stored bf16.
//
// WG = 256 threads (8 waves), output tile 128(M) x 64(N), K chunks of 32,
// double-buffered LDS (one barrier per chunk). Wave w owns rows [16w,16w+16)
// and all 64 columns: 4 accumulators, A fragment reused 4x.
//  - A chunk (f32 x) goes through VGPRs (needs f32->bf16 cvt) into LDS.
//  - B chunk (bf16 weights) is staged with GLOBAL_LOAD_ASYNC_TO_LDS_B128:
//    direct memory->LDS DMA under ASYNCcnt, no VGPR round-trip.
// Grid: blockIdx.x = N-tile (fast) so WGs sharing an M-stripe are launch-
// adjacent -> x rows are L2 hits after the first read (x ~ read once).
// ---------------------------------------------------------------------------
__global__ __launch_bounds__(256) void gemm_xp_kernel(
    const float*  __restrict__ x,    // [B*T, D] row-major (flattened [B,T,D])
    const bf16_t* __restrict__ wIh,  // [H, D] bf16 (row n = W_ih[n][:])
    const float*  __restrict__ bias, // [H]
    bf16_t*       __restrict__ xp)   // [T, B, H] bf16
{
    __shared__ bf16_t As[2][128][40];   // 40: rows stay 16B-aligned, padded
    __shared__ bf16_t Bs[2][64][40];

    const int tid  = threadIdx.x;
    const int lane = tid & 31;
    const int wav  = tid >> 5;               // 0..7 -> M sub-stripe
    const int n0   = blockIdx.x * 64;        // N fast
    const int m0   = blockIdx.y * 128;
    const int half = lane >> 4;
    const int l15  = lane & 15;

    // A-stage map: 256 thr x 4 rows x 4 f32 = 128x32 chunk
    const int sa_row = tid >> 3;             // 0..31 (+32j)
    const int sa_k   = (tid & 7) * 4;        // 0..28
    // B-stage map: 256 thr x 8 bf16 (16B) = 64x32 chunk
    const int sb_n   = tid >> 2;             // 0..63
    const int sb_k   = (tid & 3) * 8;        // 0,8,16,24

    const int arow = wav * 16 + l15;

    v8f acc[4];
#pragma unroll
    for (int j = 0; j < 4; ++j)
        acc[j] = (v8f){0.f, 0.f, 0.f, 0.f, 0.f, 0.f, 0.f, 0.f};

    auto stage = [&](int k0, int buf) {
        // B: async DMA memory -> LDS (bf16, no conversion needed)
        async_copy_b128(&Bs[buf][sb_n][sb_k],
                        wIh + (size_t)(n0 + sb_n) * Dd + k0 + sb_k);
        // A: load f32, convert to bf16 in VALU, store to LDS
#pragma unroll
        for (int j = 0; j < 4; ++j) {
            const int r = sa_row + 32 * j;
            const float4 v = *reinterpret_cast<const float4*>(
                x + (size_t)(m0 + r) * Dd + k0 + sa_k);
            As[buf][r][sa_k + 0] = (bf16_t)v.x;
            As[buf][r][sa_k + 1] = (bf16_t)v.y;
            As[buf][r][sa_k + 2] = (bf16_t)v.z;
            As[buf][r][sa_k + 3] = (bf16_t)v.w;
        }
    };

    stage(0, 0);
    wait_asynccnt0();
    __syncthreads();

    int buf = 0;
    for (int k0 = 0; k0 < Dd; k0 += 32) {
        if (k0 + 32 < Dd) stage(k0 + 32, buf ^ 1);  // prefetch into other buffer

        // A fragment: lane half 0 -> K{0..7,16..23}, half 1 -> K{8..15,24..31}
        const v16bf af = cat8(
            *reinterpret_cast<const v8bf*>(&As[buf][arow][half * 8]),
            *reinterpret_cast<const v8bf*>(&As[buf][arow][16 + half * 8]));
#pragma unroll
        for (int j = 0; j < 4; ++j) {
            const int bn = j * 16 + l15;     // B lane = column, K{8h..8h+7,16+8h..16+8h+7}
            const v16bf bfr = cat8(
                *reinterpret_cast<const v8bf*>(&Bs[buf][bn][half * 8]),
                *reinterpret_cast<const v8bf*>(&Bs[buf][bn][16 + half * 8]));
            acc[j] = wmma_bf16(af, bfr, acc[j]);
        }
        wait_asynccnt0();                    // async B writes LDS-complete
        __syncthreads();                     // single barrier per chunk
        buf ^= 1;
    }

    // epilogue: C lane l, vgpr r -> (m = 8*half + r, n = l&15); add bias, bf16
#pragma unroll
    for (int j = 0; j < 4; ++j) {
        const int ncol = n0 + j * 16 + l15;
        const float bv = bias[ncol];
#pragma unroll
        for (int r = 0; r < 8; ++r) {
            const int mm = m0 + wav * 16 + half * 8 + r;  // flat row in [B*T]
            const int b  = mm >> 9;                       // / T (=512)
            const int t  = mm & (Tt - 1);                 // % T
            xp[((size_t)t * Bb + b) * Hh + ncol] = (bf16_t)(acc[j][r] + bv);
        }
    }
}

// ---------------------------------------------------------------------------
// Phase 2: one timestep  h_out = tanh(xp_t + h_in @ W_hh^T)
//
// Grid (B/16, H/128), 128 threads = 4 waves. Each wave owns a 16x32 tile:
// 2 accumulators -> A fragment reused 2x and two independent wmma dep-chains
// per K-iteration to hide matrix-pipe latency. W_hh (2MB bf16) and h (256KB)
// are L2-resident; fragments load straight from global, no LDS, no barriers.
// ---------------------------------------------------------------------------
__global__ __launch_bounds__(128) void rnn_step_kernel(
    const bf16_t* __restrict__ h_in,   // [B, H] bf16
    const bf16_t* __restrict__ wHh,    // [H, H] bf16 (row n = W_hh[n][:])
    const bf16_t* __restrict__ xp_t,   // [B, H] bf16 (this timestep)
    bf16_t*       __restrict__ h_out,  // [B, H] bf16
    float*        __restrict__ out_f32)// d_out on last step, else nullptr
{
    const int tid  = threadIdx.x;
    const int lane = tid & 31;
    const int wav  = tid >> 5;               // 0..3
    const int m0   = blockIdx.x * 16;
    const int n0   = blockIdx.y * 128 + wav * 32;
    const int half = lane >> 4;
    const int l15  = lane & 15;

    const bf16_t* aptr  = h_in + (size_t)(m0 + l15) * Hh;
    const bf16_t* bptr0 = wHh + (size_t)(n0 + l15) * Hh;
    const bf16_t* bptr1 = wHh + (size_t)(n0 + 16 + l15) * Hh;

    v8f acc0 = {0.f, 0.f, 0.f, 0.f, 0.f, 0.f, 0.f, 0.f};
    v8f acc1 = {0.f, 0.f, 0.f, 0.f, 0.f, 0.f, 0.f, 0.f};

#pragma unroll 4
    for (int k0 = 0; k0 < Hh; k0 += 32) {
        const v16bf af = cat8(ldv8(aptr + k0 + half * 8),
                              ldv8(aptr + k0 + 16 + half * 8));
        const v16bf bf0 = cat8(ldv8(bptr0 + k0 + half * 8),
                               ldv8(bptr0 + k0 + 16 + half * 8));
        const v16bf bf1 = cat8(ldv8(bptr1 + k0 + half * 8),
                               ldv8(bptr1 + k0 + 16 + half * 8));
        acc0 = wmma_bf16(af, bf0, acc0);
        acc1 = wmma_bf16(af, bf1, acc1);
    }

#pragma unroll
    for (int j = 0; j < 2; ++j) {
        const v8f& a = j ? acc1 : acc0;
        const int ncol = n0 + j * 16 + l15;
#pragma unroll
        for (int r = 0; r < 8; ++r) {
            const int m = m0 + half * 8 + r;
            const size_t idx = (size_t)m * Hh + ncol;
            const float v = tanhf(a[r] + (float)xp_t[idx]);
            h_out[idx] = (bf16_t)v;
            if (out_f32) out_f32[idx] = v;
        }
    }
}

// ---------------------------------------------------------------------------
// Launch: prep (weight/bias convert, h0=0) -> big GEMM -> 512 step launches.
// Workspace layout (bytes):
//   [0, 2MB)        W_ih bf16
//   [2MB, 4MB)      W_hh bf16
//   [4MB, +4KB)     bias f32
//   [4MB+64KB, ...) h ping-pong bf16 (2 x 256KB)
//   [8MB, 136MB)    xp bf16 [T][B][H]
// ---------------------------------------------------------------------------
extern "C" void kernel_launch(void* const* d_in, const int* in_sizes, int n_in,
                              void* d_out, int out_size, void* d_ws, size_t ws_size,
                              hipStream_t stream) {
    (void)in_sizes; (void)n_in; (void)out_size; (void)ws_size;

    const float* x    = (const float*)d_in[0];  // [B,T,D]
    const float* W_ih = (const float*)d_in[1];  // [H,D]
    const float* W_hh = (const float*)d_in[2];  // [H,H]
    const float* b_ih = (const float*)d_in[3];  // [H]
    const float* b_hh = (const float*)d_in[4];  // [H]

    char* ws = (char*)d_ws;
    bf16_t* wIh  = (bf16_t*)(ws);
    bf16_t* wHh  = (bf16_t*)(ws + (size_t)2 * 1024 * 1024);
    float*  bias = (float*) (ws + (size_t)4 * 1024 * 1024);
    bf16_t* h0   = (bf16_t*)(ws + (size_t)4 * 1024 * 1024 + 64 * 1024);
    bf16_t* h1   = h0 + (size_t)Bb * Hh;
    bf16_t* xp   = (bf16_t*)(ws + (size_t)8 * 1024 * 1024);

    // prep: bf16 weight copies, fused bias, zeroed h0
    cvt_f32_to_bf16<<<(Hh * Dd + 255) / 256, 256, 0, stream>>>(W_ih, wIh, Hh * Dd);
    cvt_f32_to_bf16<<<(Hh * Hh + 255) / 256, 256, 0, stream>>>(W_hh, wHh, Hh * Hh);
    bias_sum_kernel<<<(Hh + 255) / 256, 256, 0, stream>>>(b_ih, b_hh, bias, Hh);
    zero_bf16_kernel<<<(Bb * Hh + 255) / 256, 256, 0, stream>>>(h0, Bb * Hh);

    // phase 1: xp = x @ W_ih^T + bias   (M = B*T = 65536)
    dim3 g1(Hh / 64, (Bb * Tt) / 128);      // N-tile fastest -> x L2 reuse
    gemm_xp_kernel<<<g1, 256, 0, stream>>>(x, wIh, bias, xp);

    // phase 2: 512 serial steps (graph-captured launches)
    bf16_t* hb[2] = {h0, h1};
    dim3 g2(Bb / 16, Hh / 128);
    for (int t = 0; t < Tt; ++t) {
        rnn_step_kernel<<<g2, 128, 0, stream>>>(
            hb[t & 1], wHh, xp + (size_t)t * Bb * Hh, hb[(t + 1) & 1],
            (t == Tt - 1) ? (float*)d_out : nullptr);
    }
}
